// SAAM_45646912422425
// MI455X (gfx1250) — matrix operand order repaired and checked by reading, not compiled, not run on hardware
//
#include <hip/hip_runtime.h>

typedef __attribute__((ext_vector_type(16))) _Float16 v16h;
typedef __attribute__((ext_vector_type(8)))  float    v8f;

#define B_  8
#define C_  64
#define HW  4096
#define NT  128      // n-tile per block (8 waves x 16)
#define KC  64       // K chunk (two f16 WMMA depths)
#define ROWU 36      // LDS row stride in dwords (32 data + 4 pad, 144B keeps 16B align)

__device__ __forceinline__ unsigned pack2h(float a, float b) {
  union { _Float16 h[2]; unsigned u; } r;
  r.h[0] = (_Float16)a; r.h[1] = (_Float16)b;
  return r.u;
}

// ---------------- Kernel 1: avg[b,n] = mean_c x[b,c,n] ----------------
__global__ void saam_avg_kernel(const float* __restrict__ x, float* __restrict__ avg) {
  int idx = blockIdx.x * blockDim.x + threadIdx.x;   // 0 .. 32767
  int b = idx >> 12, n = idx & (HW - 1);
  const float* p = x + (size_t)b * C_ * HW + n;
  float s = 0.f;
  #pragma unroll
  for (int c = 0; c < C_; ++c) s += p[(size_t)c * HW];
  avg[idx] = s * (1.0f / C_);
}

// ---------------- Kernel 2: per-batch max/min of avg ----------------
__global__ void saam_minmax_kernel(const float* __restrict__ avg, float* __restrict__ mm) {
  __shared__ float smax[256], smin[256];
  int b = blockIdx.x;
  const float* a = avg + b * HW;
  float mx = -3.4e38f, mn = 3.4e38f;
  for (int i = threadIdx.x; i < HW; i += 256) {
    float v = a[i];
    mx = fmaxf(mx, v); mn = fminf(mn, v);
  }
  smax[threadIdx.x] = mx; smin[threadIdx.x] = mn;
  __syncthreads();
  for (int s = 128; s > 0; s >>= 1) {
    if (threadIdx.x < s) {
      smax[threadIdx.x] = fmaxf(smax[threadIdx.x], smax[threadIdx.x + s]);
      smin[threadIdx.x] = fminf(smin[threadIdx.x], smin[threadIdx.x + s]);
    }
    __syncthreads();
  }
  if (threadIdx.x == 0) { mm[b * 2] = smax[0]; mm[b * 2 + 1] = smin[0]; }
}

// ---- Kernel 3: fused softmax(rank-1 scores) x V, double-buffered WMMA pipeline ----
__global__ void __launch_bounds__(256)
saam_attn_kernel(const float* __restrict__ x, const float* __restrict__ avg,
                 const float* __restrict__ mm, float* __restrict__ out) {
  __shared__ __align__(16) unsigned s_v[2][C_ * ROWU];  // V chunks, packed f16 pairs
  __shared__ float s_am[2][KC];                         // avg[m0 .. m0+63]

  const int tid  = threadIdx.x;
  const int lane = tid & 31;
  const int wave = tid >> 5;
  const int nloc = lane & 15;      // N within 16x16 tile (B/C/D layouts)
  const int g    = lane >> 4;      // half-wave group (K-half select)

  const int b  = blockIdx.y;
  const int ng = blockIdx.x * NT + wave * 16 + nloc;    // global n for this lane

  const float avg_n = avg[b * HW + ng];
  const float gmax  = mm[b * 2], gmin = mm[b * 2 + 1];
  const float rmax  = (avg_n >= 0.f) ? avg_n * gmax : avg_n * gmin; // exact rowmax

  const float* Vb = x + (size_t)b * C_ * HW;

  // cooperative loader: each thread loads 16 floats of one V row per chunk
  const int lc = tid >> 2;           // row c (0..63)
  const int lk = (tid & 3) * 16;     // k offset within chunk

  v8f acc0 = {}, acc1 = {}, acc2 = {}, acc3 = {};
  float zpart = 0.f;

  // ---- preload chunk 0 into buffer 0 ----
  {
    const float* src = Vb + (size_t)lc * HW + lk;
    float4 f0 = *(const float4*)(src);
    float4 f1 = *(const float4*)(src + 4);
    float4 f2 = *(const float4*)(src + 8);
    float4 f3 = *(const float4*)(src + 12);
    uint4 qa = make_uint4(pack2h(f0.x, f0.y), pack2h(f0.z, f0.w),
                          pack2h(f1.x, f1.y), pack2h(f1.z, f1.w));
    uint4 qb = make_uint4(pack2h(f2.x, f2.y), pack2h(f2.z, f2.w),
                          pack2h(f3.x, f3.y), pack2h(f3.z, f3.w));
    *(uint4*)&s_v[0][lc * ROWU + (lk >> 1)]     = qa;
    *(uint4*)&s_v[0][lc * ROWU + (lk >> 1) + 4] = qb;
    if (tid < KC) s_am[0][tid] = avg[b * HW + tid];
  }
  __syncthreads();

  int phase = 0;
  for (int m0 = 0; m0 < HW; m0 += KC) {
    const bool havenext = (m0 + KC) < HW;        // uniform
    uint4 qa = make_uint4(0, 0, 0, 0), qb = qa;
    float amn = 0.f;
    if (havenext) {
      // issue next chunk's global loads early; convert to packed f16 in regs
      const float* src = Vb + (size_t)lc * HW + (m0 + KC) + lk;
      if (m0 + 2 * KC < HW) __builtin_prefetch(src + KC, 0, 3);  // near-cache prefetch
      float4 f0 = *(const float4*)(src);
      float4 f1 = *(const float4*)(src + 4);
      float4 f2 = *(const float4*)(src + 8);
      float4 f3 = *(const float4*)(src + 12);
      qa = make_uint4(pack2h(f0.x, f0.y), pack2h(f0.z, f0.w),
                      pack2h(f1.x, f1.y), pack2h(f1.z, f1.w));
      qb = make_uint4(pack2h(f2.x, f2.y), pack2h(f2.z, f2.w),
                      pack2h(f3.x, f3.y), pack2h(f3.z, f3.w));
      if (tid < KC) amn = avg[b * HW + m0 + KC + tid];
    }

    // B fragments: P[m0+k, n] = exp(avg_n*avg_m - rowmax), two 32x16 f16 tiles.
    // B k order used here (contiguous, same as A): lane = N (0..15 | 16..31), K = g*16 + 2v + h.
    union { unsigned u[8]; v16h v; } bf0, bf1;
    const float* am = s_am[phase];
    #pragma unroll
    for (int v = 0; v < 8; ++v) {
      float p0 = __expf(fmaf(avg_n, am[g * 16 + 2 * v],          -rmax));
      float p1 = __expf(fmaf(avg_n, am[g * 16 + 2 * v + 1],      -rmax));
      float p2 = __expf(fmaf(avg_n, am[32 + g * 16 + 2 * v],     -rmax));
      float p3 = __expf(fmaf(avg_n, am[32 + g * 16 + 2 * v + 1], -rmax));
      zpart += (p0 + p1) + (p2 + p3);
      bf0.u[v] = pack2h(p0, p1);
      bf1.u[v] = pack2h(p2, p3);
    }

    // A fragments: 16x32 tile in this code's contiguous k order (same as B), M = lane%16,
    // VGPR v: K = Kbase + g*16 + v*2 + h  -> two b128 LDS reads each.
    const unsigned* sv = s_v[phase];
    #pragma unroll
    for (int t = 0; t < 4; ++t) {
      const int M = t * 16 + nloc;
      union { uint4 q[2]; v16h v; } af;
      v8f* ap = (t == 0) ? &acc0 : (t == 1) ? &acc1 : (t == 2) ? &acc2 : &acc3;
      af.q[0] = *(const uint4*)&sv[M * ROWU + g * 8];
      af.q[1] = *(const uint4*)&sv[M * ROWU + 4 + g * 8];
      *ap = __builtin_amdgcn_wmma_f32_16x16x32_f16(
          false, af.v, false, bf0.v, (short)0, *ap, false, false);
      af.q[0] = *(const uint4*)&sv[M * ROWU + 16 + g * 8];
      af.q[1] = *(const uint4*)&sv[M * ROWU + 20 + g * 8];
      *ap = __builtin_amdgcn_wmma_f32_16x16x32_f16(
          false, af.v, false, bf1.v, (short)0, *ap, false, false);
    }

    // stage next chunk into the other buffer; single barrier per iteration
    if (havenext) {
      unsigned* dv = s_v[phase ^ 1];
      *(uint4*)&dv[lc * ROWU + (lk >> 1)]     = qa;
      *(uint4*)&dv[lc * ROWU + (lk >> 1) + 4] = qb;
      if (tid < KC) s_am[phase ^ 1][tid] = amn;
    }
    __syncthreads();
    phase ^= 1;
  }

  // Z[n]: lanes l and l^16 cover complementary K-halves of the same n
  float z  = zpart + __shfl_xor(zpart, 16, 32);
  float rz = 1.0f / z;

  // C/D f32 16x16 layout: lane, reg r -> M = r + 8*(lane/16), N = lane%16
  float* ob = out + (size_t)b * C_ * HW;
  #pragma unroll
  for (int t = 0; t < 4; ++t) {
    const v8f a = (t == 0) ? acc0 : (t == 1) ? acc1 : (t == 2) ? acc2 : acc3;
    #pragma unroll
    for (int r = 0; r < 8; ++r) {
      int c = t * 16 + r + 8 * g;
      ob[(size_t)c * HW + ng] = a[r] * rz;
    }
  }
}

extern "C" void kernel_launch(void* const* d_in, const int* in_sizes, int n_in,
                              void* d_out, int out_size, void* d_ws, size_t ws_size,
                              hipStream_t stream) {
  const float* x = (const float*)d_in[0];
  float* out = (float*)d_out;
  float* avg = (float*)d_ws;        // 32768 floats
  float* mm  = avg + B_ * HW;       // 16 floats: {max,min} x 8 batches

  saam_avg_kernel<<<(B_ * HW) / 256, 256, 0, stream>>>(x, avg);
  saam_minmax_kernel<<<B_, 256, 0, stream>>>(avg, mm);
  dim3 grid(HW / NT, B_);
  saam_attn_kernel<<<grid, 256, 0, stream>>>(x, avg, mm, out);
}
